// HierarchicalAttentionMechanism_26310969655904
// MI455X (gfx1250) — compile-verified, matrix operand order checked by reading, not run on hardware
//
#include <hip/hip_runtime.h>
#include <hip/hip_bf16.h>
#include <math.h>

// ---------------------------------------------------------------------------
// HierarchicalAttentionMechanism for MI455X (gfx1250), wave32 + WMMA bf16.
// GEMMs: v_wmma_f32_16x16x32_bf16, 128x128 block tile, BK=64, double-buffered
// LDS with gfx1250 async global->LDS copies overlapping WMMA.
// ---------------------------------------------------------------------------

typedef __attribute__((ext_vector_type(8)))  __bf16 v8bf;
typedef __attribute__((ext_vector_type(16))) __bf16 v16bf;
typedef __attribute__((ext_vector_type(8)))  float  v8f;
typedef int v4i_ __attribute__((vector_size(16)));

#if defined(__gfx1250__) && __has_builtin(__builtin_amdgcn_global_load_async_to_lds_b128)
#define USE_ASYNC_LDS 1
#endif

__device__ __forceinline__ void cp16(const __bf16* g, __bf16* l) {
#ifdef USE_ASYNC_LDS
  __builtin_amdgcn_global_load_async_to_lds_b128(
      (__attribute__((address_space(1))) v4i_*)(void*)g,
      (__attribute__((address_space(3))) v4i_*)(void*)l, 0, 0);
#else
  *(v8bf*)l = *(const v8bf*)g;
#endif
}

__device__ __forceinline__ void async_wait0() {
#ifdef USE_ASYNC_LDS
#if __has_builtin(__builtin_amdgcn_s_wait_asynccnt)
  __builtin_amdgcn_s_wait_asynccnt(0);
#else
  asm volatile("s_wait_asynccnt 0x0" ::: "memory");
#endif
#endif
}

__device__ __forceinline__ v16bf load_frag(const __bf16* rowPtr, int kgrp) {
  // ISA 16-bit fragment striping: lane(0-15) holds K[kgrp..kgrp+7] then
  // K[kgrp+16..kgrp+23]; kgrp = 0 for lanes 0-15, 8 for lanes 16-31.
  v8bf lo = *(const v8bf*)(rowPtr + kgrp);
  v8bf hi = *(const v8bf*)(rowPtr + kgrp + 16);
  return __builtin_shufflevector(lo, hi, 0,1,2,3,4,5,6,7,8,9,10,11,12,13,14,15);
}

__device__ __forceinline__ v8f wmma_bf16(v16bf a, v16bf b, v8f c) {
  return __builtin_amdgcn_wmma_f32_16x16x32_bf16(false, a, false, b, (short)0,
                                                 c, false, false);
}

// ---------------------------------------------------------------------------
// Positional encoding add: xpe = x + PE, plus bf16 shadow copy.
// ---------------------------------------------------------------------------
__global__ __launch_bounds__(256) void pe_add_kernel(
    const float* __restrict__ x, float* __restrict__ xpe,
    __bf16* __restrict__ xpeb, int S, int D)
{
  size_t i = (size_t)blockIdx.x * 256 + threadIdx.x;
  int d = (int)(i % D);
  int s = (int)((i / D) % S);
  float freq = __expf((float)(d & ~1) * (-9.210340371976184f / (float)D));
  float ang = (float)s * freq;
  float pe = (d & 1) ? __cosf(ang) : __sinf(ang);
  float v = x[i] + pe;
  xpe[i] = v;
  xpeb[i] = (__bf16)v;
}

__global__ __launch_bounds__(256) void cvt_bf16_kernel(
    const float* __restrict__ a, __bf16* __restrict__ o, size_t n)
{
  size_t i = (size_t)blockIdx.x * 256 + threadIdx.x;
  if (i < n) o[i] = (__bf16)a[i];
}

// gather strided rows: xs[m,:] = xpe[(b*S + (m%Sl)*stride), :]  (8-half chunks)
__global__ __launch_bounds__(256) void gather_kernel(
    const __bf16* __restrict__ xpeb, __bf16* __restrict__ xs,
    int Sl, int stride, int S, int D)
{
  size_t i = (size_t)blockIdx.x * 256 + threadIdx.x;   // chunk index
  int cpr = D / 8;
  int m = (int)(i / cpr), c = (int)(i % cpr);
  int b = m / Sl, s = (m % Sl) * stride;
  ((v8bf*)xs)[i] = ((const v8bf*)xpeb)[(size_t)(b * S + s) * cpr + c];
}

// ---------------------------------------------------------------------------
// Generic GEMM: C[M,N] = A[M,K](bf16) @ W[N,K]^T(bf16) + bias[N]
// Block tile 128x128, BK=64, 256 threads = 8 waves in a 4(M) x 2(N) grid,
// each wave owns a 32x64 sub-tile = 2x4 WMMA accumulators.
// Double-buffered LDS; tile k+1 staged (async DMA if available) while WMMA
// consumes tile k. Requires M%128==0, N%128==0, K%64==0 (all call sites).
// ---------------------------------------------------------------------------
__global__ __launch_bounds__(256) void gemm_bf16_kernel(
    const __bf16* __restrict__ A, const __bf16* __restrict__ W,
    const float* __restrict__ bias,
    float* __restrict__ Cf, __bf16* __restrict__ Cb, int outBf,
    int M, int N, int K)
{
  __shared__ __bf16 sA[2][128][72];   // 64 + 8 pad halves: 144B rows, 16B align
  __shared__ __bf16 sB[2][128][72];
  const int t = threadIdx.x;
  const int lane = t & 31, wave = t >> 5;
  const int wm = wave >> 1, wn = wave & 1;
  const int lrow = lane & 15;
  const int hi8 = (lane >> 4) * 8;           // kgrp / row-half offset
  const size_t aBase = (size_t)blockIdx.y * 128 * (size_t)K;
  const size_t bBase = (size_t)blockIdx.x * 128 * (size_t)K;
  const int colBase = blockIdx.x * 128 + wn * 64;
  const int rowBase = blockIdx.y * 128 + wm * 32;

  // staging geometry: 128 rows x 8 chunks(8 halves) per matrix = 1024 chunks;
  // 256 threads -> 4 chunks each, rows rA + {0,32,64,96}.
  const int rA = t >> 3;          // 0..31
  const int cA = (t & 7) * 8;     // 0..56

  v8f acc[2][4];
  #pragma unroll
  for (int ni = 0; ni < 4; ni++) {
    float bv = bias[colBase + ni * 16 + lrow];
    #pragma unroll
    for (int mi = 0; mi < 2; mi++)
      #pragma unroll
      for (int r = 0; r < 8; r++) acc[mi][ni][r] = bv;
  }

  // ---- prologue: stage tile 0 into buffer 0 ----
#ifdef USE_ASYNC_LDS
  #pragma unroll
  for (int c = 0; c < 4; c++) {
    cp16(A + aBase + (size_t)(rA + c * 32) * K + cA, &sA[0][rA + c * 32][cA]);
    cp16(W + bBase + (size_t)(rA + c * 32) * K + cA, &sB[0][rA + c * 32][cA]);
  }
  async_wait0();
#else
  {
    v8bf ra[4], rb[4];
    #pragma unroll
    for (int c = 0; c < 4; c++) {
      ra[c] = *(const v8bf*)(A + aBase + (size_t)(rA + c * 32) * K + cA);
      rb[c] = *(const v8bf*)(W + bBase + (size_t)(rA + c * 32) * K + cA);
    }
    #pragma unroll
    for (int c = 0; c < 4; c++) {
      *(v8bf*)&sA[0][rA + c * 32][cA] = ra[c];
      *(v8bf*)&sB[0][rA + c * 32][cA] = rb[c];
    }
  }
#endif
  __syncthreads();

  int buf = 0;
  for (int kt = 0; kt < K; kt += 64) {
    const bool hasNext = (kt + 64) < K;

    // ---- stage tile k+1 into buf^1 (overlaps with WMMA below) ----
#ifdef USE_ASYNC_LDS
    if (hasNext) {
      #pragma unroll
      for (int c = 0; c < 4; c++) {
        cp16(A + aBase + (size_t)(rA + c * 32) * K + kt + 64 + cA,
             &sA[buf ^ 1][rA + c * 32][cA]);
        cp16(W + bBase + (size_t)(rA + c * 32) * K + kt + 64 + cA,
             &sB[buf ^ 1][rA + c * 32][cA]);
      }
    }
#else
    v8bf ra[4], rb[4];
    if (hasNext) {
      #pragma unroll
      for (int c = 0; c < 4; c++) {
        ra[c] = *(const v8bf*)(A + aBase + (size_t)(rA + c * 32) * K + kt + 64 + cA);
        rb[c] = *(const v8bf*)(W + bBase + (size_t)(rA + c * 32) * K + kt + 64 + cA);
      }
    }
#endif

    // ---- compute current tile: 2 K-steps x (2x4) WMMA ----
    #pragma unroll
    for (int ks = 0; ks < 2; ks++) {
      v16bf af[2], bfr[4];
      #pragma unroll
      for (int mi = 0; mi < 2; mi++)
        af[mi] = load_frag(&sA[buf][wm * 32 + mi * 16 + lrow][ks * 32], hi8);
      #pragma unroll
      for (int ni = 0; ni < 4; ni++)
        bfr[ni] = load_frag(&sB[buf][wn * 64 + ni * 16 + lrow][ks * 32], hi8);
      #pragma unroll
      for (int mi = 0; mi < 2; mi++)
        #pragma unroll
        for (int ni = 0; ni < 4; ni++)
          acc[mi][ni] = wmma_bf16(af[mi], bfr[ni], acc[mi][ni]);
    }

    // ---- publish tile k+1 ----
#ifdef USE_ASYNC_LDS
    if (hasNext) async_wait0();
#else
    if (hasNext) {
      #pragma unroll
      for (int c = 0; c < 4; c++) {
        *(v8bf*)&sA[buf ^ 1][rA + c * 32][cA] = ra[c];
        *(v8bf*)&sB[buf ^ 1][rA + c * 32][cA] = rb[c];
      }
    }
#endif
    __syncthreads();
    buf ^= 1;
  }

  #pragma unroll
  for (int mi = 0; mi < 2; mi++)
    #pragma unroll
    for (int ni = 0; ni < 4; ni++) {
      int col = colBase + ni * 16 + lrow;
      #pragma unroll
      for (int r = 0; r < 8; r++) {
        int row = rowBase + mi * 16 + hi8 + r;
        size_t o = (size_t)row * N + col;
        if (outBf) Cb[o] = (__bf16)acc[mi][ni][r];
        else       Cf[o] = acc[mi][ni][r];
      }
    }
}

// ---------------------------------------------------------------------------
// Attention: one workgroup per (head, batch). qkv: (B*Sl, 3072) bf16.
// out: (B*Sl, 1024) bf16 head-concat. Sl in {64,32,16,8}.
// ---------------------------------------------------------------------------
__global__ __launch_bounds__(128) void attn_kernel(
    const __bf16* __restrict__ qkv, __bf16* __restrict__ outp, int Sl)
{
  const int h = blockIdx.x, b = blockIdx.y;
  const int D3 = 3072;
  __shared__ __bf16 sQ[64][72];
  __shared__ __bf16 sK[64][72];
  __shared__ __bf16 sVt[64][72];   // transposed: sVt[d][s]
  __shared__ __bf16 sP[64][72];
  const int t = threadIdx.x;

  // zero-fill (pads rows/cols beyond Sl for masking & K-padding)
  for (int i = t; i < 64 * 72; i += 128) {
    ((__bf16*)sQ)[i]  = (__bf16)0.0f;
    ((__bf16*)sK)[i]  = (__bf16)0.0f;
    ((__bf16*)sVt)[i] = (__bf16)0.0f;
    ((__bf16*)sP)[i]  = (__bf16)0.0f;
  }
  __syncthreads();

  for (int i = t; i < Sl * 8; i += 128) {
    int s = i >> 3, dc = (i & 7) * 8;
    size_t base = (size_t)(b * Sl + s) * D3 + h * 64 + dc;
    v8bf q = *(const v8bf*)(qkv + base);
    v8bf k = *(const v8bf*)(qkv + base + 1024);
    v8bf v = *(const v8bf*)(qkv + base + 2048);
    *(v8bf*)&sQ[s][dc] = q;
    *(v8bf*)&sK[s][dc] = k;
    #pragma unroll
    for (int j = 0; j < 8; j++) sVt[dc + j][s] = v[j];
  }
  __syncthreads();

  const int lane = t & 31, w = t >> 5;
  const int lrow = lane & 15, hi8 = (lane >> 4) * 8;
  const int St = (Sl + 15) & ~15;
  const int Nt = St >> 4;
  const int Kp = (St < 32) ? 32 : St;
  const bool act = (w * 16) < St;

  v8f accS[4];
  if (act) {
    v16bf qa0 = load_frag(&sQ[w * 16 + lrow][0], hi8);   // K 0..31
    v16bf qa1 = load_frag(&sQ[w * 16 + lrow][32], hi8);  // K 32..63
    for (int j = 0; j < Nt; j++) {
      v16bf kb0 = load_frag(&sK[j * 16 + lrow][0], hi8);
      v16bf kb1 = load_frag(&sK[j * 16 + lrow][32], hi8);
      v8f c = {};
      c = wmma_bf16(qa0, kb0, c);
      c = wmma_bf16(qa1, kb1, c);
      accS[j] = c * 0.125f;                 // 1/sqrt(64)
    }
    // mask invalid key columns
    for (int j = 0; j < Nt; j++) {
      int col = j * 16 + lrow;
      if (col >= Sl)
        #pragma unroll
        for (int r = 0; r < 8; r++) accS[j][r] = -1e30f;
    }
    // row-wise softmax (rows live across lanes 0-15 / 16-31 half-waves)
    for (int r = 0; r < 8; r++) {
      float mx = -1e30f;
      for (int j = 0; j < Nt; j++) mx = fmaxf(mx, accS[j][r]);
      for (int msk = 8; msk >= 1; msk >>= 1)
        mx = fmaxf(mx, __shfl_xor(mx, msk, 16));
      float sum = 0.0f;
      for (int j = 0; j < Nt; j++) {
        float e = __expf(accS[j][r] - mx);
        accS[j][r] = e;
        sum += e;
      }
      for (int msk = 8; msk >= 1; msk >>= 1)
        sum += __shfl_xor(sum, msk, 16);
      float inv = 1.0f / sum;
      for (int j = 0; j < Nt; j++) accS[j][r] *= inv;
    }
    // probs -> LDS (plain row-major tile; reloaded in A-frag layout)
    for (int j = 0; j < Nt; j++)
      #pragma unroll
      for (int r = 0; r < 8; r++)
        sP[w * 16 + hi8 + r][j * 16 + lrow] = (__bf16)accS[j][r];
  }
  __syncthreads();

  if (act) {
    v8f accO[4] = {};
    for (int kt = 0; kt < Kp; kt += 32) {
      v16bf pa = load_frag(&sP[w * 16 + lrow][kt], hi8);
      #pragma unroll
      for (int jd = 0; jd < 4; jd++) {
        v16bf vb = load_frag(&sVt[jd * 16 + lrow][kt], hi8);
        accO[jd] = wmma_bf16(pa, vb, accO[jd]);
      }
    }
    #pragma unroll
    for (int jd = 0; jd < 4; jd++) {
      int col = h * 64 + jd * 16 + lrow;
      #pragma unroll
      for (int r = 0; r < 8; r++) {
        int row = w * 16 + hi8 + r;
        if (row < Sl)
          outp[(size_t)(b * Sl + row) * 1024 + col] = (__bf16)accO[jd][r];
      }
    }
  }
}

// ---------------------------------------------------------------------------
// LayerNorm + exact GELU epilogue, with row scatter into cat buffer.
// row m -> out row (m/Sl)*S + (m%Sl)*stride, column offset colOff.
// ---------------------------------------------------------------------------
__global__ __launch_bounds__(256) void ln_gelu_kernel(
    const float* __restrict__ in, const float* __restrict__ g,
    const float* __restrict__ bta, __bf16* __restrict__ outp,
    int N, int Sl, int stride, int S, int outStride, int colOff)
{
  __shared__ float red[256];
  __shared__ float s_mean, s_rstd;
  const int m = blockIdx.x, t = threadIdx.x;
  const float* row = in + (size_t)m * N;
  float sum = 0.0f, sq = 0.0f;
  for (int n = t; n < N; n += 256) { float v = row[n]; sum += v; sq += v * v; }
  red[t] = sum; __syncthreads();
  for (int s2 = 128; s2 > 0; s2 >>= 1) { if (t < s2) red[t] += red[t + s2]; __syncthreads(); }
  if (t == 0) s_mean = red[0] / (float)N;
  __syncthreads();
  red[t] = sq; __syncthreads();
  for (int s2 = 128; s2 > 0; s2 >>= 1) { if (t < s2) red[t] += red[t + s2]; __syncthreads(); }
  if (t == 0) {
    float var = red[0] / (float)N - s_mean * s_mean;
    s_rstd = rsqrtf(var + 1e-5f);
  }
  __syncthreads();
  const float mean = s_mean, rstd = s_rstd;
  const int bb = m / Sl, ss = (m % Sl) * stride;
  const size_t orow = ((size_t)bb * S + ss) * (size_t)outStride + colOff;
  for (int n = t; n < N; n += 256) {
    float v = (row[n] - mean) * rstd * g[n] + bta[n];
    v = 0.5f * v * (1.0f + erff(v * 0.70710678118f));
    outp[orow + n] = (__bf16)v;
  }
}

// Fill non-strided rows of cat for level>0: gelu(LN(pb)) constant vector.
__global__ __launch_bounds__(256) void fill_const_kernel(
    const float* __restrict__ pb, const float* __restrict__ pg,
    const float* __restrict__ pbeta, __bf16* __restrict__ cat,
    int Sl, int stride, int S, int colOff, int N)
{
  __shared__ float red[256];
  __shared__ float s_mean, s_rstd;
  const int t = threadIdx.x;
  const int nFill = S - Sl;
  const int b = blockIdx.x / nFill, r = blockIdx.x % nFill;
  const int s = (r / (stride - 1)) * stride + (r % (stride - 1)) + 1;

  float sum = 0.0f, sq = 0.0f;
  for (int n = t; n < N; n += 256) { float v = pb[n]; sum += v; sq += v * v; }
  red[t] = sum; __syncthreads();
  for (int s2 = 128; s2 > 0; s2 >>= 1) { if (t < s2) red[t] += red[t + s2]; __syncthreads(); }
  if (t == 0) s_mean = red[0] / (float)N;
  __syncthreads();
  red[t] = sq; __syncthreads();
  for (int s2 = 128; s2 > 0; s2 >>= 1) { if (t < s2) red[t] += red[t + s2]; __syncthreads(); }
  if (t == 0) {
    float var = red[0] / (float)N - s_mean * s_mean;
    s_rstd = rsqrtf(var + 1e-5f);
  }
  __syncthreads();
  const float mean = s_mean, rstd = s_rstd;
  const size_t orow = ((size_t)b * S + s) * 4096 + colOff;
  for (int n = t; n < N; n += 256) {
    float v = (pb[n] - mean) * rstd * pg[n] + pbeta[n];
    v = 0.5f * v * (1.0f + erff(v * 0.70710678118f));
    cat[orow + n] = (__bf16)v;
  }
}

// Final: out = LN(h2)*g2 + b2 + xpe   (f32 output)
__global__ __launch_bounds__(256) void final_kernel(
    const float* __restrict__ in, const float* __restrict__ g,
    const float* __restrict__ bta, const float* __restrict__ xpe,
    float* __restrict__ outp, int N)
{
  __shared__ float red[256];
  __shared__ float s_mean, s_rstd;
  const int m = blockIdx.x, t = threadIdx.x;
  const float* row = in + (size_t)m * N;
  float sum = 0.0f, sq = 0.0f;
  for (int n = t; n < N; n += 256) { float v = row[n]; sum += v; sq += v * v; }
  red[t] = sum; __syncthreads();
  for (int s2 = 128; s2 > 0; s2 >>= 1) { if (t < s2) red[t] += red[t + s2]; __syncthreads(); }
  if (t == 0) s_mean = red[0] / (float)N;
  __syncthreads();
  red[t] = sq; __syncthreads();
  for (int s2 = 128; s2 > 0; s2 >>= 1) { if (t < s2) red[t] += red[t + s2]; __syncthreads(); }
  if (t == 0) {
    float var = red[0] / (float)N - s_mean * s_mean;
    s_rstd = rsqrtf(var + 1e-5f);
  }
  __syncthreads();
  const float mean = s_mean, rstd = s_rstd;
  for (int n = t; n < N; n += 256) {
    float v = (row[n] - mean) * rstd * g[n] + bta[n];
    outp[(size_t)m * N + n] = v + xpe[(size_t)m * N + n];
  }
}

// ---------------------------------------------------------------------------
// Host orchestration
// ---------------------------------------------------------------------------
extern "C" void kernel_launch(void* const* d_in, const int* in_sizes, int n_in,
                              void* d_out, int out_size, void* d_ws, size_t ws_size,
                              hipStream_t stream) {
  (void)in_sizes; (void)n_in; (void)out_size; (void)ws_size;
  const int B = 256, S = 64, D = 1024, Hh = 16, Lh = 4;
  const int M0 = B * S;                      // 16384
  const int D3 = 3 * D, D4 = 4 * D, D2 = 2 * D;

  const float* x    = (const float*)d_in[0];
  const float* Wqkv = (const float*)d_in[1];
  const float* bqkv = (const float*)d_in[2];
  const float* Wo   = (const float*)d_in[3];
  const float* bo   = (const float*)d_in[4];
  const float* pW   = (const float*)d_in[5];
  const float* pb   = (const float*)d_in[6];
  const float* pg   = (const float*)d_in[7];
  const float* pbeta= (const float*)d_in[8];
  const float* fW1  = (const float*)d_in[9];
  const float* fb1  = (const float*)d_in[10];
  const float* g1   = (const float*)d_in[11];
  const float* b1   = (const float*)d_in[12];
  const float* fW2  = (const float*)d_in[13];
  const float* fb2  = (const float*)d_in[14];
  const float* g2   = (const float*)d_in[15];
  const float* b2   = (const float*)d_in[16];
  float* out = (float*)d_out;

  // ---- workspace carve ----
  char* p = (char*)d_ws;
  auto carve = [&](size_t bytes) { char* r = p; p += (bytes + 255) & ~(size_t)255; return r; };
  float*  xpe   = (float*) carve((size_t)M0 * D * 4);       // 64 MB (residual)
  __bf16* xpeb  = (__bf16*)carve((size_t)M0 * D * 2);       // 32 MB
  __bf16* wqkvb = (__bf16*)carve((size_t)Lh * D3 * D * 2);  // 24 MB
  __bf16* wob   = (__bf16*)carve((size_t)Lh * D * D * 2);   //  8 MB
  __bf16* pwb   = (__bf16*)carve((size_t)Lh * D * D * 2);   //  8 MB
  __bf16* fw1b  = (__bf16*)carve((size_t)D2 * D4 * 2);      // 16 MB
  __bf16* fw2b  = (__bf16*)carve((size_t)D * D2 * 2);       //  4 MB
  __bf16* cat   = (__bf16*)carve((size_t)M0 * D4 * 2);      // 128 MB
  char*   arena = carve((size_t)256 * 1024 * 1024);         // 256 MB (aliased)

  // phase A (per-level) views
  __bf16* xs      = (__bf16*)(arena);
  __bf16* qkv     = (__bf16*)(arena + (size_t) 32 * 1024 * 1024);
  __bf16* attnout = (__bf16*)(arena + (size_t)128 * 1024 * 1024);
  __bf16* ybuf    = (__bf16*)(arena + (size_t)160 * 1024 * 1024);
  float*  hproj   = (float*) (arena + (size_t)192 * 1024 * 1024);
  // phase B (FFN) views
  float*  h1f = (float*) (arena);
  __bf16* h1b = (__bf16*)(arena + (size_t)128 * 1024 * 1024);
  float*  h2f = (float*) (arena + (size_t)192 * 1024 * 1024);

  // ---- setup: PE add + weight conversion ----
  {
    size_t n = (size_t)M0 * D;
    pe_add_kernel<<<dim3((unsigned)(n / 256)), 256, 0, stream>>>(x, xpe, xpeb, S, D);
  }
  auto cvt = [&](const float* src, __bf16* dst, size_t n) {
    cvt_bf16_kernel<<<dim3((unsigned)((n + 255) / 256)), 256, 0, stream>>>(src, dst, n);
  };
  cvt(Wqkv, wqkvb, (size_t)Lh * D3 * D);
  cvt(Wo,   wob,   (size_t)Lh * D * D);
  cvt(pW,   pwb,   (size_t)Lh * D * D);
  cvt(fW1,  fw1b,  (size_t)D2 * D4);
  cvt(fW2,  fw2b,  (size_t)D * D2);

  auto gemm = [&](const __bf16* A, const __bf16* W, const float* bias,
                  float* Cf, __bf16* Cb, int outBf, int M, int N, int K) {
    dim3 g((unsigned)(N / 128), (unsigned)(M / 128));
    gemm_bf16_kernel<<<g, 256, 0, stream>>>(A, W, bias, Cf, Cb, outBf, M, N, K);
  };

  // ---- per-level pipeline ----
  for (int lev = 0; lev < Lh; lev++) {
    const int stride = 1 << lev;
    const int Sl = S >> lev;
    const int Ml = B * Sl;

    const __bf16* xsrc = xpeb;
    if (lev > 0) {
      size_t chunks = (size_t)Ml * D / 8;
      gather_kernel<<<dim3((unsigned)(chunks / 256)), 256, 0, stream>>>(
          xpeb, xs, Sl, stride, S, D);
      xsrc = xs;
    }
    // QKV: (Ml, 3072) = xs @ Wqkv^T + bqkv  -> bf16
    gemm(xsrc, wqkvb + (size_t)lev * D3 * D, bqkv + lev * D3,
         nullptr, qkv, 1, Ml, D3, D);
    // attention per (head, batch)
    attn_kernel<<<dim3(Hh, B), 128, 0, stream>>>(qkv, attnout, Sl);
    // Wo: (Ml, 1024) -> bf16
    gemm(attnout, wob + (size_t)lev * D * D, bo + lev * D,
         nullptr, ybuf, 1, Ml, D, D);
    // pW: (Ml, 1024) -> f32 (pre-LN)
    gemm(ybuf, pwb + (size_t)lev * D * D, pb + lev * D,
         hproj, nullptr, 0, Ml, D, D);
    // LN + GELU, scatter strided rows into cat at column lev*D
    ln_gelu_kernel<<<dim3((unsigned)Ml), 256, 0, stream>>>(
        hproj, pg + lev * D, pbeta + lev * D, cat, D, Sl, stride, S, D4, lev * D);
    // constant rows (zero attention rows -> gelu(LN(pb)))
    if (lev > 0) {
      int nFill = B * (S - Sl);
      fill_const_kernel<<<dim3((unsigned)nFill), 256, 0, stream>>>(
          pb + lev * D, pg + lev * D, pbeta + lev * D, cat, Sl, stride, S, lev * D, D);
    }
  }

  // ---- FFN ----
  gemm(cat, fw1b, fb1, h1f, nullptr, 0, M0, D2, D4);             // 275 GFLOP
  ln_gelu_kernel<<<dim3((unsigned)M0), 256, 0, stream>>>(
      h1f, g1, b1, h1b, D2, S, 1, S, D2, 0);                     // identity map
  gemm(h1b, fw2b, fb2, h2f, nullptr, 0, M0, D, D2);
  final_kernel<<<dim3((unsigned)M0), 256, 0, stream>>>(h2f, g2, b2, xpe, out, D);
}
